// MultiHeadAttention_75754633167270
// MI455X (gfx1250) — matrix operand order repaired and checked by reading, not compiled, not run on hardware
//
#include <hip/hip_runtime.h>

#define BB 4
#define SS 2048
#define DD 1024
#define HH 16
#define DHH 64

// all four GEMMs in this model share the same shape
#define GEMM_M 8192   // BB*SS
#define GEMM_N 1024   // DD
#define GEMM_K 1024   // DD

typedef unsigned short ushort_t;
typedef __bf16 bf16_t;
typedef bf16_t bf16x16 __attribute__((ext_vector_type(16)));
typedef float  floatx8 __attribute__((ext_vector_type(8)));

#if __has_builtin(__builtin_amdgcn_global_load_async_to_lds_b128) && \
    __has_builtin(__builtin_amdgcn_s_wait_asynccnt)
#define ASYNC_OK 1
#else
#define ASYNC_OK 0
#endif

// GCC-style vector-of-4-int matching the async builtin's pointee type,
// with address-space-qualified flavors.
typedef int v4i_g __attribute__((vector_size(16)));
typedef __attribute__((address_space(1))) v4i_g AS1_v4i;
typedef __attribute__((address_space(3))) v4i_g AS3_v4i;

#if ASYNC_OK
#define ASYNC_CP128(gp, lp) \
  __builtin_amdgcn_global_load_async_to_lds_b128((AS1_v4i*)(gp), (AS3_v4i*)(lp), 0, 0)
#define ASYNC_WAIT(n) __builtin_amdgcn_s_wait_asynccnt(n)
#else
#define ASYNC_CP128(gp, lp) (*(uint4*)(lp) = *(const uint4*)(gp))
#define ASYNC_WAIT(n)
#endif

union FragU { bf16x16 v; uint4 q[2]; };

__device__ __forceinline__ floatx8 zero8() {
  floatx8 z = {0.f, 0.f, 0.f, 0.f, 0.f, 0.f, 0.f, 0.f};
  return z;
}

// float -> bf16, round-to-nearest-even
__device__ __forceinline__ ushort_t f2bf(float f) {
  unsigned int u = __float_as_uint(f);
  unsigned int r = u + 0x7FFFu + ((u >> 16) & 1u);
  return (ushort_t)(r >> 16);
}

// ---------------------------------------------------------------------------
// A-fragment: 16x32 bf16 tile, row-major, ld elements per row.
// ISA 7.12.2: lanes 0-15 row M=lane, elems 0..7 = K[kh*8..kh*8+7],
//             elems 8..15 = K[16+kh*8 ..], kh = lane>>4.
// ---------------------------------------------------------------------------
__device__ __forceinline__ bf16x16 load_frag_a(const ushort_t* base, int ld) {
  const int lane = threadIdx.x & 31;
  const int row  = lane & 15;
  const int kh   = (lane >> 4) & 1;
  const ushort_t* r = base + (size_t)row * ld + kh * 8;
  FragU f;
  f.q[0] = *(const uint4*)(r);
  f.q[1] = *(const uint4*)(r + 16);
  return f.v;
}

// ---------------------------------------------------------------------------
// B-fragment: 32(K) x 16(N) bf16 tile stored N-major: mat[n][k], ld = row
// stride (elements). N=lane&15, kh=lane>>4: elems 0..7 = K[kh*8..kh*8+7],
// elems 8..15 = K[16+kh*8 ..] (same order as A).
// ---------------------------------------------------------------------------
__device__ __forceinline__ bf16x16 load_frag_b(const ushort_t* base, int ld) {
  const int lane = threadIdx.x & 31;
  const int col  = lane & 15;
  const int kh   = (lane >> 4) & 1;
  const ushort_t* r = base + (size_t)col * ld + kh * 8;
  FragU f;
  f.q[0] = *(const uint4*)(r);
  f.q[1] = *(const uint4*)(r + 16);
  return f.v;
}

__device__ __forceinline__ floatx8 wmma_bf16(bf16x16 a, bf16x16 b, floatx8 c) {
  return __builtin_amdgcn_wmma_f32_16x16x32_bf16(
      /*neg_a=*/false, a, /*neg_b=*/false, b,
      /*c_mod=*/(short)0, c, /*reuse_a=*/false, /*reuse_b=*/false);
}

// ---------------------------------------------------------------------------
// fp32 -> bf16 elementwise conversion (grid-stride)
// ---------------------------------------------------------------------------
__global__ void cvt_f32_bf16(const float* __restrict__ src,
                             ushort_t* __restrict__ dst, int n) {
  int i = blockIdx.x * blockDim.x + threadIdx.x;
  int stride = gridDim.x * blockDim.x;
  for (; i < n; i += stride) dst[i] = f2bf(src[i]);
}

// ---------------------------------------------------------------------------
// GEMM: out(M,N) = A(M,K row-major) * B(N,K N-major)^T + bias
// Fixed shape 8192x1024x1024. Block = 256 threads = 8 waves, block tile
// 128x128 double-buffered in LDS via GLOBAL_LOAD_ASYNC_TO_LDS_B128/ASYNCcnt.
// Wave tile 64x32 (waves 2Mx4N) -> 8 WMMA per K-step, 4x A + 2x B reuse from
// LDS, 64 FLOP per staged global byte at block level. Steady state unrolled
// by 2 (buffer parity is compile-time const), last K-step peeled.
// mode 0: bf16 out, (B,H,S,DH) layout      (Q, K projections)
// mode 1: bf16 out, (B,H,DH,S) layout      (V projection, pre-transposed)
// mode 2: fp32 out, row-major (M,N)        (final output projection)
// ---------------------------------------------------------------------------
#define GBM 128
#define GBN 128
#define GBK 32
#define NSTEP (GEMM_K / GBK)  // 32
#define LDST 40  // padded LDS row stride (elements): 80B, 16B-aligned, bank-spread

__global__ void __launch_bounds__(256)
gemm_bf16(const ushort_t* __restrict__ A, const ushort_t* __restrict__ Bm,
          const float* __restrict__ bias, void* __restrict__ outp, int mode) {
  __shared__ __align__(16) ushort_t sA[2][GBM * LDST];
  __shared__ __align__(16) ushort_t sB[2][GBN * LDST];

  const int tid  = threadIdx.x;
  const int lane = tid & 31;
  const int wid  = tid >> 5;
  const int wy   = wid >> 2;          // 0..1  (M)
  const int wx   = wid & 3;           // 0..3  (N)
  const int bm0  = blockIdx.y * GBM;
  const int bn0  = blockIdx.x * GBN;

  // Hoisted per-thread staging addresses: tile is 128 rows x 4 chunks of 16B;
  // thread handles rows r0 and r0+64, chunk c0.
  const int r0 = tid >> 2;             // 0..63
  const int c0 = (tid & 3) * 8;        // element offset within row
  const ushort_t* gA0 = A  + (size_t)(bm0 + r0) * GEMM_K + c0;
  const ushort_t* gA1 = gA0 + (size_t)64 * GEMM_K;
  const ushort_t* gB0 = Bm + (size_t)(bn0 + r0) * GEMM_K + c0;
  const ushort_t* gB1 = gB0 + (size_t)64 * GEMM_K;
  const int l0 = r0 * LDST + c0;
  const int l1 = l0 + 64 * LDST;

  // stage K-slice kk (elements) into LDS buffer pb: 4 async b128 per thread
  auto stage = [&](int kk, int pb) {
    ASYNC_CP128(gA0 + kk, &sA[pb][l0]);
    ASYNC_CP128(gA1 + kk, &sA[pb][l1]);
    ASYNC_CP128(gB0 + kk, &sB[pb][l0]);
    ASYNC_CP128(gB1 + kk, &sB[pb][l1]);
  };

  floatx8 acc[4][2];
#pragma unroll
  for (int i = 0; i < 4; ++i)
#pragma unroll
    for (int j = 0; j < 2; ++j) acc[i][j] = zero8();

  // 8 WMMA from LDS buffer pb (compile-time const at every call site)
  auto compute = [&](int pb) {
    bf16x16 af[4], bfr[2];
#pragma unroll
    for (int t = 0; t < 4; ++t)
      af[t] = load_frag_a(&sA[pb][(wy * 64 + t * 16) * LDST], LDST);
#pragma unroll
    for (int t = 0; t < 2; ++t)
      bfr[t] = load_frag_b(&sB[pb][(wx * 32 + t * 16) * LDST], LDST);
#pragma unroll
    for (int i = 0; i < 4; ++i)
#pragma unroll
      for (int j = 0; j < 2; ++j)
        acc[i][j] = wmma_bf16(af[i], bfr[j], acc[i][j]);
  };

  // ---- async double-buffered main loop (branch-free steady state) ----
  stage(0, 0);
  for (int s = 0; s < NSTEP - 2; s += 2) {
    const int kk = s * GBK;
    stage(kk + GBK, 1);        // slice s+1 -> buf1 (8 outstanding)
    ASYNC_WAIT(4);             // slice s landed; s+1 still in flight
    __syncthreads();
    compute(0);
    __syncthreads();
    stage(kk + 2 * GBK, 0);    // slice s+2 -> buf0
    ASYNC_WAIT(4);
    __syncthreads();
    compute(1);
    __syncthreads();
  }
  // peeled tail: steps NSTEP-2 (buf0) and NSTEP-1 (buf1)
  stage((NSTEP - 1) * GBK, 1);
  ASYNC_WAIT(4);
  __syncthreads();
  compute(0);
  __syncthreads();
  ASYNC_WAIT(0);
  __syncthreads();
  compute(1);

  // ---- epilogue ----
  const int col  = lane & 15;
  const int rowb = (lane >> 4) * 8;
#pragma unroll
  for (int tm = 0; tm < 4; ++tm) {
#pragma unroll
    for (int tn = 0; tn < 2; ++tn) {
      const int n = bn0 + wx * 32 + tn * 16 + col;
      const float bv = bias ? bias[n] : 0.0f;
#pragma unroll
      for (int v = 0; v < 8; ++v) {
        const int m = bm0 + wy * 64 + tm * 16 + rowb + v;
        const float val = acc[tm][tn][v] + bv;
        if (mode == 2) {
          ((float*)outp)[(size_t)m * GEMM_N + n] = val;
        } else {
          const int b = m / SS, s = m % SS;
          const int h = n / DHH, d = n % DHH;
          const size_t idx = (mode == 0)
              ? (((size_t)(b * HH + h) * SS + s) * DHH + d)   // (B,H,S,DH)
              : (((size_t)(b * HH + h) * DHH + d) * SS + s);  // (B,H,DH,S)
          ((ushort_t*)outp)[idx] = f2bf(val);
        }
      }
    }
  }
}

// ---------------------------------------------------------------------------
// Flash attention: one wave per 16 query rows; stream keys in chunks of 32.
// Q,K in (B,H,S,DH) bf16; V in (B,H,DH,S) bf16 (loads as B-fragment).
// scores = (mask + Q*K^T) / 8 (faithful quirk: mask added before scaling).
// Online softmax, P staged through LDS to re-shape into an A-fragment.
// ---------------------------------------------------------------------------
__global__ void __launch_bounds__(128)
flash_attn(const ushort_t* __restrict__ Q, const ushort_t* __restrict__ Kc,
           const ushort_t* __restrict__ Vt, const float* __restrict__ mask,
           ushort_t* __restrict__ O) {
  __shared__ __align__(16) ushort_t plds[4][16 * 32];

  const int lane = threadIdx.x & 31;
  const int wid  = threadIdx.x >> 5;
  const int b = blockIdx.z, h = blockIdx.y;
  const int q0 = blockIdx.x * 64 + wid * 16;

  const ushort_t* Qh = Q  + ((size_t)(b * HH + h) * SS) * DHH;
  const ushort_t* Kh = Kc + ((size_t)(b * HH + h) * SS) * DHH;
  const ushort_t* Vh = Vt + ((size_t)(b * HH + h) * DHH) * SS;
  const float*    mb = mask + (size_t)b * SS * SS;

  const bf16x16 aq0 = load_frag_a(Qh + (size_t)q0 * DHH, DHH);
  const bf16x16 aq1 = load_frag_a(Qh + (size_t)q0 * DHH + 32, DHH);

  floatx8 accO[4] = {zero8(), zero8(), zero8(), zero8()};
  float mrow[8], lrow[8];
#pragma unroll
  for (int v = 0; v < 8; ++v) { mrow[v] = -3.0e38f; lrow[v] = 0.0f; }

  const int col  = lane & 15;
  const int rowb = (lane >> 4) * 8;
  ushort_t* pl = plds[wid];

  for (int kc = 0; kc < SS; kc += 32) {
    floatx8 s0 = zero8(), s1 = zero8();
    s0 = wmma_bf16(aq0, load_frag_b(Kh + (size_t)kc * DHH, DHH), s0);
    s0 = wmma_bf16(aq1, load_frag_b(Kh + (size_t)kc * DHH + 32, DHH), s0);
    s1 = wmma_bf16(aq0, load_frag_b(Kh + (size_t)(kc + 16) * DHH, DHH), s1);
    s1 = wmma_bf16(aq1, load_frag_b(Kh + (size_t)(kc + 16) * DHH + 32, DHH), s1);

    float scale[8];
#pragma unroll
    for (int v = 0; v < 8; ++v) {
      const int q = q0 + rowb + v;
      float x0 = (s0[v] + mb[(size_t)q * SS + kc + col]) * 0.125f;
      float x1 = (s1[v] + mb[(size_t)q * SS + kc + 16 + col]) * 0.125f;
      float cm = fmaxf(x0, x1);
      cm = fmaxf(cm, __shfl_xor(cm, 1, 32));
      cm = fmaxf(cm, __shfl_xor(cm, 2, 32));
      cm = fmaxf(cm, __shfl_xor(cm, 4, 32));
      cm = fmaxf(cm, __shfl_xor(cm, 8, 32));
      const float mnew = fmaxf(mrow[v], cm);
      const float sc = __expf(mrow[v] - mnew);
      const float p0 = __expf(x0 - mnew);
      const float p1 = __expf(x1 - mnew);
      float rs = p0 + p1;
      rs += __shfl_xor(rs, 1, 32);
      rs += __shfl_xor(rs, 2, 32);
      rs += __shfl_xor(rs, 4, 32);
      rs += __shfl_xor(rs, 8, 32);
      lrow[v] = lrow[v] * sc + rs;
      mrow[v] = mnew;
      scale[v] = sc;
      pl[(rowb + v) * 32 + col]      = f2bf(p0);
      pl[(rowb + v) * 32 + col + 16] = f2bf(p1);
    }

#pragma unroll
    for (int t = 0; t < 4; ++t)
#pragma unroll
      for (int v = 0; v < 8; ++v) accO[t][v] *= scale[v];

    // wave-local LDS RAW: wait for DS stores, no barrier needed (one wave)
    asm volatile("s_wait_dscnt 0x0" ::: "memory");
    const bf16x16 ap = load_frag_a(pl, 32);

#pragma unroll
    for (int t = 0; t < 4; ++t)
      accO[t] = wmma_bf16(ap, load_frag_b(Vh + (size_t)(t * 16) * SS + kc, SS),
                          accO[t]);
  }

#pragma unroll
  for (int v = 0; v < 8; ++v) {
    const int q = q0 + rowb + v;
    const float inv = 1.0f / lrow[v];
#pragma unroll
    for (int t = 0; t < 4; ++t) {
      const int d = t * 16 + col;
      O[((size_t)b * SS + q) * DD + h * DHH + d] = f2bf(accO[t][v] * inv);
    }
  }
}

// ---------------------------------------------------------------------------
extern "C" void kernel_launch(void* const* d_in, const int* in_sizes, int n_in,
                              void* d_out, int out_size, void* d_ws,
                              size_t ws_size, hipStream_t stream) {
  const float* x    = (const float*)d_in[0];
  const float* mask = (const float*)d_in[1];
  const float* WQw  = (const float*)d_in[2];
  const float* WQb  = (const float*)d_in[3];
  const float* WKw  = (const float*)d_in[4];
  const float* WVw  = (const float*)d_in[5];
  const float* WVb  = (const float*)d_in[6];
  const float* WOw  = (const float*)d_in[7];
  const float* WOb  = (const float*)d_in[8];

  const size_t nX = (size_t)BB * SS * DD;  // 8M elements
  const size_t nW = (size_t)DD * DD;       // 1M elements

  ushort_t* xbf = (ushort_t*)d_ws;
  ushort_t* wqb = xbf + nX;
  ushort_t* wkb = wqb + nW;
  ushort_t* wvb = wkb + nW;
  ushort_t* wob = wvb + nW;
  ushort_t* Qb  = wob + nW;   // (B,H,S,DH)
  ushort_t* Kb  = Qb + nX;    // (B,H,S,DH)
  ushort_t* Vtb = Kb + nX;    // (B,H,DH,S)
  ushort_t* Ob  = Vtb + nX;   // (B,S,D)
  // total: 44M bf16 = 88 MB of workspace (fits in the 192 MB L2)

  cvt_f32_bf16<<<2048, 256, 0, stream>>>(x, xbf, (int)nX);
  cvt_f32_bf16<<<1024, 256, 0, stream>>>(WQw, wqb, (int)nW);
  cvt_f32_bf16<<<1024, 256, 0, stream>>>(WKw, wkb, (int)nW);
  cvt_f32_bf16<<<1024, 256, 0, stream>>>(WVw, wvb, (int)nW);
  cvt_f32_bf16<<<1024, 256, 0, stream>>>(WOw, wob, (int)nW);

  dim3 gg(GEMM_N / GBN, GEMM_M / GBM);  // (8, 64)
  gemm_bf16<<<gg, 256, 0, stream>>>(xbf, wqb, WQb, Qb, 0);
  gemm_bf16<<<gg, 256, 0, stream>>>(xbf, wkb, nullptr, Kb, 0);
  gemm_bf16<<<gg, 256, 0, stream>>>(xbf, wvb, WVb, Vtb, 1);

  dim3 ga(SS / 64, HH, BB);  // (32, 16, 4)
  flash_attn<<<ga, 128, 0, stream>>>(Qb, Kb, Vtb, mask, Ob);

  gemm_bf16<<<gg, 256, 0, stream>>>(Ob, wob, WOb, d_out, 2);
}
